// VanillaAttention_10995116277779
// MI455X (gfx1250) — matrix operand order repaired and checked by reading, not compiled, not run on hardware
//
#include <hip/hip_runtime.h>

typedef __attribute__((ext_vector_type(16))) __bf16 v16bf;
typedef __attribute__((ext_vector_type(8)))  __bf16 v8bf;
typedef __attribute__((ext_vector_type(2)))  __bf16 v2bf;
typedef __attribute__((ext_vector_type(8)))  float  v8f;
typedef __attribute__((ext_vector_type(4)))  float  v4f;

#define BHROWS 32         // B*H
#define SEQ    2048
#define DIM    128
#define BM     128        // Q rows per workgroup (8 waves x 16)
#define BN     32         // K/V rows per iteration

__launch_bounds__(256, 1)
__global__ void fa_fwd_bf16wmma(const float* __restrict__ Q,
                                const float* __restrict__ K,
                                const float* __restrict__ V,
                                float* __restrict__ O)
{
    // double-buffered tiles: 2 * (8KB + 8KB) = 32KB of the 320KB WGP LDS
    __shared__ __align__(32) __bf16 sK [2][BN * DIM];   // [key][d] row-major (A operand)
    __shared__ __align__(32) __bf16 sVt[2][DIM * BN];   // [d][key] transposed (B operand)

    const int tid   = threadIdx.x;
    const int lane  = tid & 31;
    const int wave  = tid >> 5;
    const int half  = lane >> 4;
    const int l16   = lane & 15;
    const int half8 = half * 8;

    const long base  = (long)blockIdx.y * SEQ * DIM;
    const int  qrow0 = blockIdx.x * BM + wave * 16;

    const float sl = 0.088388347648318447f * 1.4426950408889634f; // (1/sqrt(D))*log2(e)

    // ---- Q as WMMA B-operand, pre-scaled by sl: lane: N=q-row=l16, K(d)=t*32+half*16+0..15 ----
    v16bf qb[4];
    {
        const float* qp = Q + base + (long)(qrow0 + l16) * DIM + half * 16;
#pragma unroll
        for (int t = 0; t < 4; ++t) {
#pragma unroll
            for (int j = 0; j < 4; ++j) {
                const v4f f = *(const v4f*)(qp + t * 32 + j * 4);
#pragma unroll
                for (int i = 0; i < 4; ++i) qb[t][j * 4 + i] = (__bf16)(f[i] * sl);
            }
        }
    }

    v8f   o[8] = {};                  // O: 8 d-tiles, C layout (M=q-row, N=d)
    float mrow = -1.0e30f;            // per-lane running stats of q-row l16 (exp2 domain)
    float lrow = 0.0f;

    // staging map: each thread owns 2 key rows x 8 d-values (enables packed stores)
    const int sRow = (tid >> 4) * 2;       // even key row 0..30
    const int sCol = (tid & 15) * 8;       // d offset 0..120
    const float* kp0 = K + base + (long)sRow * DIM + sCol;
    const float* vp0 = V + base + (long)sRow * DIM + sCol;

    v4f kr[4], vr[4];   // [0..1]: row sRow, [2..3]: row sRow+1
#pragma unroll
    for (int j = 0; j < 2; ++j) {
        kr[j]     = *(const v4f*)(kp0 + j * 4);
        kr[2 + j] = *(const v4f*)(kp0 + DIM + j * 4);
        vr[j]     = *(const v4f*)(vp0 + j * 4);
        vr[2 + j] = *(const v4f*)(vp0 + DIM + j * 4);
    }

    // commit registers -> LDS buffer pp (K row-major b128, V transposed packed b32)
    auto stage = [&](int pp) {
        v8bf k0, k1;
#pragma unroll
        for (int i = 0; i < 4; ++i) {
            k0[i] = (__bf16)kr[0][i];  k0[4 + i] = (__bf16)kr[1][i];
            k1[i] = (__bf16)kr[2][i];  k1[4 + i] = (__bf16)kr[3][i];
        }
        *(v8bf*)&sK[pp][sRow * DIM + sCol]       = k0;
        *(v8bf*)&sK[pp][(sRow + 1) * DIM + sCol] = k1;
#pragma unroll
        for (int j = 0; j < 8; ++j) {
            v2bf p;
            p[0] = (__bf16)((j < 4) ? vr[0][j] : vr[1][j - 4]);
            p[1] = (__bf16)((j < 4) ? vr[2][j] : vr[3][j - 4]);
            *(v2bf*)&sVt[pp][(sCol + j) * BN + sRow] = p;   // sRow even -> 4B aligned
        }
    };

    stage(0);
    __syncthreads();

    int pp = 0;
    for (int kb = 0; kb < SEQ; kb += BN, pp ^= 1) {
        const bool more = (kb + BN) < SEQ;
        if (more) {      // pipeline: fetch next tile into registers now
            const float* kp = kp0 + (long)(kb + BN) * DIM;
            const float* vp = vp0 + (long)(kb + BN) * DIM;
#pragma unroll
            for (int j = 0; j < 2; ++j) {
                kr[j]     = *(const v4f*)(kp + j * 4);
                kr[2 + j] = *(const v4f*)(kp + DIM + j * 4);
                vr[j]     = *(const v4f*)(vp + j * 4);
                vr[2 + j] = *(const v4f*)(vp + DIM + j * 4);
            }
            if (kb + 2 * BN < SEQ) {   // warm L2 for the tile after that
                __builtin_prefetch(kp0 + (long)(kb + 2 * BN) * DIM, 0, 3);
                __builtin_prefetch(vp0 + (long)(kb + 2 * BN) * DIM, 0, 3);
            }
        }

        // ---- S^T = K * Q^T : two 16x16 tiles (M=key half c, N=q-row) ----
        v8f st[2];
#pragma unroll
        for (int c = 0; c < 2; ++c) {
            v8f acc = {};
            const int key = c * 16 + l16;            // A row M = local key
#pragma unroll
            for (int t = 0; t < 4; ++t) {
                v16bf ka;
                const v8bf a0 = *(const v8bf*)&sK[pp][key * DIM + t * 32 + half * 16];
                const v8bf a1 = *(const v8bf*)&sK[pp][key * DIM + t * 32 + half * 16 + 8];
#pragma unroll
                for (int i = 0; i < 8; ++i) { ka[i] = a0[i]; ka[8 + i] = a1[i]; }
                acc = __builtin_amdgcn_wmma_f32_16x16x32_bf16(false, ka, false, qb[t],
                                                              (short)0, acc, false, false);
            }
            st[c] = acc;
        }

        // ---- online softmax: q-row lives in-lane (16 vals) + partner lane (xor 16) ----
        // (logits already scaled: Q was pre-multiplied by (1/sqrt(D))*log2(e))
        float x[16];
#pragma unroll
        for (int v = 0; v < 8; ++v) { x[v] = st[0][v]; x[8 + v] = st[1][v]; }
        float mx = x[0];
#pragma unroll
        for (int i = 1; i < 16; ++i) mx = fmaxf(mx, x[i]);
        mx = fmaxf(mx, __shfl_xor(mx, 16, 32));
        const float mn = fmaxf(mrow, mx);
        const float a  = __builtin_exp2f(mrow - mn);

        v16bf pa;                       // P lands directly in WMMA A-layout registers
        float rs = 0.0f;
#pragma unroll
        for (int i = 0; i < 16; ++i) {
            const float p = __builtin_exp2f(x[i] - mn);
            rs += p;
            pa[i] = (__bf16)p;
        }
        rs += __shfl_xor(rs, 16, 32);
        lrow = lrow * a + rs;
        mrow = mn;

        // ---- rescale O: row M = v + 8*half needs alpha of q-row M (lives in lane M) ----
#pragma unroll
        for (int v = 0; v < 8; ++v) {
            const float av = __shfl(a, half8 + v, 32);
#pragma unroll
            for (int dc = 0; dc < 8; ++dc) o[dc][v] *= av;
        }

        // ---- O += P * V : 8 output d-tiles ----
#pragma unroll
        for (int dc = 0; dc < 8; ++dc) {
            const v8bf b0 = *(const v8bf*)&sVt[pp][(dc * 16 + l16) * BN + half8];
            const v8bf b1 = *(const v8bf*)&sVt[pp][(dc * 16 + l16) * BN + 16 + half8];
            v16bf vb;
#pragma unroll
            for (int i = 0; i < 8; ++i) { vb[i] = b0[i]; vb[8 + i] = b1[i]; }
            o[dc] = __builtin_amdgcn_wmma_f32_16x16x32_bf16(false, pa, false, vb,
                                                            (short)0, o[dc], false, false);
        }

        // commit next tile into the other buffer (overlaps with this tile's compute),
        // then a single barrier per iteration flips the buffers for all waves
        if (more) stage(pp ^ 1);
        __syncthreads();
    }

    // ---- normalize (broadcast per-row 1/l) and store ----
    float rl[8];
#pragma unroll
    for (int v = 0; v < 8; ++v) rl[v] = 1.0f / __shfl(lrow, half8 + v, 32);
    float* op = O + base + (long)qrow0 * DIM;
#pragma unroll
    for (int dc = 0; dc < 8; ++dc)
#pragma unroll
        for (int v = 0; v < 8; ++v)
            op[(long)(half8 + v) * DIM + dc * 16 + l16] = o[dc][v] * rl[v];
}

extern "C" void kernel_launch(void* const* d_in, const int* in_sizes, int n_in,
                              void* d_out, int out_size, void* d_ws, size_t ws_size,
                              hipStream_t stream) {
    (void)in_sizes; (void)n_in; (void)d_ws; (void)ws_size; (void)out_size;
    const float* q = (const float*)d_in[0];
    const float* k = (const float*)d_in[1];
    const float* v = (const float*)d_in[2];
    float* out = (float*)d_out;

    dim3 grid(SEQ / BM, BHROWS);   // (16, 32)
    dim3 block(256);               // 8 wave32 waves
    fa_fwd_bf16wmma<<<grid, block, 0, stream>>>(q, k, v, out);
}
